// CompressiveTransformer_2637109920396
// MI455X (gfx1250) — matrix operand order repaired and checked by reading, not compiled, not run on hardware
//
#include <hip/hip_runtime.h>
#include <math.h>

// ---------------------------------------------------------------------------
// Problem constants (compressive transformer forward)
// ---------------------------------------------------------------------------
constexpr int B_ = 4, T_ = 512, D_ = 1024, H_ = 8, DEPTH_ = 4, V_ = 32000;
constexpr int MEM_LEN_ = 512, CMEM_LEN_ = 128, RATIO_ = 4;
constexpr int DH_ = D_ / H_;                       // 128
constexpr int KV_ = CMEM_LEN_ + MEM_LEN_ + T_;     // 1152

typedef __bf16 bf16;
typedef __attribute__((ext_vector_type(16))) __bf16 v16bf;
typedef __attribute__((ext_vector_type(8)))  float  v8f;

struct U128 { unsigned x, y, z, w; };              // POD 16B chunk for LDS vector loads
union FragBF { v16bf v; U128 q[2]; };
static_assert(sizeof(v16bf) == 32, "v16bf must be 32B");

__device__ __forceinline__ bf16 f2bf(float f) {
  unsigned u = __builtin_bit_cast(unsigned, f);
  unsigned r = u + 0x7FFFu + ((u >> 16) & 1u);     // round-to-nearest-even
  unsigned short h = (unsigned short)(r >> 16);
  return __builtin_bit_cast(bf16, h);
}

// ---------------------------------------------------------------------------
// CDNA5 async copy: global -> LDS, 16 bytes per lane, tracked by ASYNCcnt.
// Low 32 bits of a generic LDS pointer are the LDS byte address (ISA 10.2).
// ---------------------------------------------------------------------------
__device__ __forceinline__ void async_ld16(const bf16* lds_dst, const bf16* gsrc) {
  unsigned l = (unsigned)(unsigned long long)(const void*)lds_dst;
  unsigned long long g = (unsigned long long)(const void*)gsrc;
  asm volatile("global_load_async_to_lds_b128 %0, %1, off"
               :: "v"(l), "v"(g) : "memory");
}
__device__ __forceinline__ void wait_async0() {
  asm volatile("s_wait_asynccnt 0x0" ::: "memory");
}

// ---------------------------------------------------------------------------
// WMMA GEMM:  C[M,N] = epi( A[M,K] * B[N,K]^T )   (bf16 operands, fp32 accum)
// 128x128x32 tile, 256 threads (8 wave32), each wave: 32x64 = 8 WMMA / K-step.
// Double-buffered LDS filled by global_load_async_to_lds_b128.
// Batched over blockIdx.z with (z/Hdiv, z%Hdiv) stride decomposition.
//
// REQUIREMENT (satisfied by every call site in this file): M % 128 == 0,
// N % 128 == 0, K % 32 == 0.  This removes all per-lane bounds guards from
// the pipelined loop (no EXEC save/restore around the async issues).
// ---------------------------------------------------------------------------
enum { EPI_NONE = 0, EPI_BIAS = 1, EPI_BIAS_GELU = 2, EPI_SCALE = 3 };

template <int EPI>
__device__ __forceinline__ float epi_apply(float v, const float* bias, int n, float scale) {
  if constexpr (EPI == EPI_SCALE) { return v * scale; }
  else if constexpr (EPI == EPI_BIAS) { return v + bias[n]; }
  else if constexpr (EPI == EPI_BIAS_GELU) {
    float t = v + bias[n];
    return 0.5f * t * (1.0f + erff(t * 0.70710678118654752f));   // exact GELU
  } else { return v; }
}

constexpr int BM = 128, BN = 128, BK = 32, BKP = 48;  // BKP: padded LDS row (96B)

template <int EPI, bool OBF>
__global__ __launch_bounds__(256) void gemm_bf16_wmma(
    const bf16* __restrict__ Abase, const bf16* __restrict__ Bbase,
    void* __restrict__ Cbase, const float* __restrict__ bias,
    int K, int lda, int ldb, int ldc,
    long sAb, long sAh, long sBb, long sBh, long sCb, long sCh,
    int Hdiv, float scale)
{
  __shared__ bf16 As[2][BM * BKP];   // 2 x 12KB
  __shared__ bf16 Bs[2][BN * BKP];   // 2 x 12KB  (N-major)

  const int z  = blockIdx.z;
  const int bz = z / Hdiv, hz = z % Hdiv;
  const long  coff = bz * sCb + hz * sCh;
  float* Cf = (float*)Cbase + coff;
  bf16*  Cb = (bf16*)Cbase + coff;

  const int tid  = threadIdx.x;
  const int lane = tid & 31;
  const int wave = tid >> 5;
  const int wm   = (wave & 3) * 32;   // wave row offset inside tile (2 x 16)
  const int wn   = (wave >> 2) * 64;  // wave col offset inside tile (4 x 16)
  const int m0   = blockIdx.y * BM;
  const int n0   = blockIdx.x * BN;

  // Per-thread staging slots are loop-invariant: row/chunk derived from tid.
  const int srow = tid >> 2;          // 0..63  (+64 on second sub-iteration)
  const int sc8  = (tid & 3) * 8;     // K-offset in halfs (16B chunks)
  const bf16* Ag = Abase + bz * sAb + hz * sAh + (long)(m0 + srow) * lda + sc8;
  const bf16* Bg = Bbase + bz * sBb + hz * sBh + (long)(n0 + srow) * ldb + sc8;
  const long aRow64 = (long)64 * lda;       // second half of the tile
  const long bRow64 = (long)64 * ldb;

  // Stage one 128x32 bf16 tile pair via async DMA: 4 x 16B chunks per thread.
  auto stage = [&](int buf, int kt) {
    bf16* la = &As[buf][srow * BKP + sc8];
    bf16* lb = &Bs[buf][srow * BKP + sc8];
    async_ld16(la,                 Ag + kt);
    async_ld16(la + 64 * BKP,      Ag + kt + aRow64);
    async_ld16(lb,                 Bg + kt);
    async_ld16(lb + 64 * BKP,      Bg + kt + bRow64);
  };

  v8f acc[2][4] = {};

  stage(0, 0);
  wait_async0();
  __syncthreads();

  int cur = 0;
  for (int kt = 0; kt < K; kt += BK) {
    if (kt + BK < K) stage(cur ^ 1, kt + BK);   // DMA next tile while computing

    const int fm = lane & 15;    // row/col inside 16-tile
    const int ks = lane >> 4;    // K-half selector per ISA fragment layout
    FragBF a[2], b[4];
#pragma unroll
    for (int i = 0; i < 2; ++i) {
      const bf16* ap = &As[cur][(wm + i * 16 + fm) * BKP];
      a[i].q[0] = *(const U128*)(ap + ks * 8);         // K ks*8..+7
      a[i].q[1] = *(const U128*)(ap + 16 + ks * 8);    // K 16+ks*8..
    }
#pragma unroll
    for (int j = 0; j < 4; ++j) {
      const bf16* bp = &Bs[cur][(wn + j * 16 + fm) * BKP + ks * 8];
      b[j].q[0] = *(const U128*)bp;                    // K ks*8..+7
      b[j].q[1] = *(const U128*)(bp + 16);
    }
#pragma unroll
    for (int i = 0; i < 2; ++i)
#pragma unroll
      for (int j = 0; j < 4; ++j)
        acc[i][j] = __builtin_amdgcn_wmma_f32_16x16x32_bf16(
            false, a[i].v, false, b[j].v, (short)0, acc[i][j], false, false);

    wait_async0();       // next tile landed (this wave's copies)
    __syncthreads();     // ... and every wave's copies; prev reads done
    cur ^= 1;
  }

  // store: lanes 0-15 -> rows +0..7, lanes 16-31 -> rows +8..15, N = lane&15
  const int cn  = n0 + wn + (lane & 15);
  const int cmr = m0 + wm + ((lane >> 4) << 3);
#pragma unroll
  for (int i = 0; i < 2; ++i) {
#pragma unroll
    for (int r = 0; r < 8; ++r) {
      const long rowoff = (long)(cmr + i * 16 + r) * ldc;
#pragma unroll
      for (int j = 0; j < 4; ++j) {
        int n = cn + j * 16;
        float v = epi_apply<EPI>(acc[i][j][r], bias, n, scale);
        if (OBF) Cb[rowoff + n] = f2bf(v);
        else     Cf[rowoff + n] = v;
      }
    }
  }
}

template <int EPI, bool OBF>
static void launch_gemm(hipStream_t s, const bf16* A, const bf16* B, void* C,
                        const float* bias, int M, int N, int K,
                        int lda, int ldb, int ldc,
                        long sAb, long sAh, long sBb, long sBh, long sCb, long sCh,
                        int Z, int Hdiv, float scale)
{
  // M, N multiples of 128; K multiple of 32 (all call sites in this file).
  dim3 grid((unsigned)(N / BN), (unsigned)(M / BM), (unsigned)Z);
  gemm_bf16_wmma<EPI, OBF><<<grid, dim3(256), 0, s>>>(
      A, B, C, bias, K, lda, ldb, ldc,
      sAb, sAh, sBb, sBh, sCb, sCh, Hdiv, scale);
}

// ---------------------------------------------------------------------------
// Operand preparation kernels
// ---------------------------------------------------------------------------
// fp32 [K][N] -> bf16 [N][K]  (tiled transpose+convert)
__global__ __launch_bounds__(256) void transpose_cvt_kernel(
    const float* __restrict__ src, bf16* __restrict__ dst, int K, int N) {
  __shared__ float tile[32][33];
  int kb = blockIdx.y * 32, nb = blockIdx.x * 32;
  int tx = threadIdx.x & 31, ty = threadIdx.x >> 5;
  for (int i = ty; i < 32; i += 8) {
    int k = kb + i, n = nb + tx;
    tile[i][tx] = (k < K && n < N) ? src[(long)k * N + n] : 0.f;
  }
  __syncthreads();
  for (int i = ty; i < 32; i += 8) {
    int n = nb + i, k = kb + tx;
    if (n < N && k < K) dst[(long)n * K + k] = f2bf(tile[tx][i]);
  }
}

__global__ void cvt_kernel(const float* __restrict__ s, bf16* __restrict__ d, long n) {
  long i = (long)blockIdx.x * blockDim.x + threadIdx.x;
  if (i < n) d[i] = f2bf(s[i]);
}

// conv_w (o,i,r) -> wpk[o][r*D + i] bf16   (B^T operand of conv-as-GEMM)
__global__ void repack_conv_kernel(const float* __restrict__ w, bf16* __restrict__ wpk, long n) {
  long idx = (long)blockIdx.x * blockDim.x + threadIdx.x;
  if (idx >= n) return;
  int o = (int)(idx >> 12);
  int rem = (int)(idx & 4095);
  int i = rem >> 2;
  int r = rem & 3;
  wpk[(long)o * (RATIO_ * D_) + (long)r * D_ + i] = f2bf(w[idx]);
}

// v half of kv -> vT[(b*H+h)][dh][j]   (B^T operand of attn*V)
__global__ void vT_kernel(const bf16* __restrict__ kv, bf16* __restrict__ vT, long n) {
  long i = (long)blockIdx.x * blockDim.x + threadIdx.x;
  if (i >= n) return;
  int z  = (int)(i / ((long)DH_ * KV_));
  int rem = (int)(i % ((long)DH_ * KV_));
  int dh = rem / KV_, j = rem % KV_;
  int b = z / H_, h = z % H_;
  vT[i] = kv[((long)b * KV_ + j) * (2 * D_) + D_ + h * DH_ + dh];
}

// ---------------------------------------------------------------------------
// Elementwise / row kernels
// ---------------------------------------------------------------------------
__global__ void embed_kernel(const int* __restrict__ tokens, const float* __restrict__ emb,
                             float* __restrict__ x, bf16* __restrict__ xb, long n) {
  long i = (long)blockIdx.x * blockDim.x + threadIdx.x;
  if (i >= n) return;
  long row = i / D_; int c = (int)(i % D_);
  float v = emb[(long)tokens[row] * D_ + c];
  x[i] = v; xb[i] = f2bf(v);
}

__global__ __launch_bounds__(256) void layernorm_kernel(
    const float* __restrict__ x, const float* __restrict__ g, const float* __restrict__ b,
    float* __restrict__ yf /*nullable*/, bf16* __restrict__ yb, int Dm) {
  __shared__ float red[2][8];
  long row = blockIdx.x;
  const float* xr = x + row * Dm;
  float s = 0.f, sq = 0.f;
  for (int c = threadIdx.x; c < Dm; c += 256) { float v = xr[c]; s += v; sq += v * v; }
  for (int o = 16; o; o >>= 1) { s += __shfl_xor(s, o, 32); sq += __shfl_xor(sq, o, 32); }
  int lane = threadIdx.x & 31, wave = threadIdx.x >> 5;
  if (lane == 0) { red[0][wave] = s; red[1][wave] = sq; }
  __syncthreads();
  if (threadIdx.x == 0) {
    float ts = 0.f, tq = 0.f;
    for (int w = 0; w < 8; ++w) { ts += red[0][w]; tq += red[1][w]; }
    float mean = ts / Dm;
    red[0][0] = mean;
    red[1][0] = rsqrtf(tq / Dm - mean * mean + 1e-5f);
  }
  __syncthreads();
  float mean = red[0][0], rstd = red[1][0];
  for (int c = threadIdx.x; c < Dm; c += 256) {
    float v = (xr[c] - mean) * rstd * g[c] + b[c];
    if (yf) yf[row * Dm + c] = v;
    yb[row * Dm + c] = f2bf(v);
  }
}

__global__ void concat_kvin_kernel(const float* __restrict__ cm, const float* __restrict__ mm,
                                   const bf16* __restrict__ xb, bf16* __restrict__ kvin, long n) {
  long i = (long)blockIdx.x * blockDim.x + threadIdx.x;
  if (i >= n) return;
  int b = (int)(i / ((long)KV_ * D_));
  int r = (int)((i / D_) % KV_);
  int c = (int)(i % D_);
  bf16 v;
  if (r < CMEM_LEN_)                 v = f2bf(cm[((long)b * CMEM_LEN_ + r) * D_ + c]);
  else if (r < CMEM_LEN_ + MEM_LEN_) v = f2bf(mm[((long)b * MEM_LEN_ + (r - CMEM_LEN_)) * D_ + c]);
  else                               v = xb[((long)b * T_ + (r - CMEM_LEN_ - MEM_LEN_)) * D_ + c];
  kvin[i] = v;
}

// dots += rel_shift(pos_raw); causal mask (j > i + KV - T -> -inf)
__global__ void shift_mask_add_kernel(float* __restrict__ dots,
                                      const float* __restrict__ posr, long n) {
  long i = (long)blockIdx.x * blockDim.x + threadIdx.x;
  if (i >= n) return;
  long rowsz = (long)T_ * KV_;
  long bh = i / rowsz;
  int r = (int)((i / KV_) % T_);
  int c = (int)(i % KV_);
  if (c > r + (KV_ - T_)) { dots[i] = -3.0e38f; return; }
  long f = (long)r * (T_ + KV_ - 1) + (T_ - 1) + c;
  int rr = (int)(f / (T_ + KV_));
  int cc = (int)(f % (T_ + KV_));
  float p = (cc < KV_) ? posr[bh * rowsz + (long)rr * KV_ + cc] : 0.f;
  dots[i] += p;
}

// row softmax (fp32 in, bf16 probabilities out)
__global__ __launch_bounds__(256) void softmax_rows_kernel(
    float* __restrict__ d, bf16* __restrict__ pb, long rows, int cols) {
  long row = (long)blockIdx.x * 8 + (threadIdx.x >> 5);
  int lane = threadIdx.x & 31;
  if (row >= rows) return;
  float* p = d + row * cols;
  bf16*  o = pb + row * cols;
  float mx = -3.4e38f;
  for (int c = lane; c < cols; c += 32) mx = fmaxf(mx, p[c]);
  for (int off = 16; off; off >>= 1) mx = fmaxf(mx, __shfl_xor(mx, off, 32));
  float sum = 0.f;
  for (int c = lane; c < cols; c += 32) { float e = __expf(p[c] - mx); p[c] = e; sum += e; }
  for (int off = 16; off; off >>= 1) sum += __shfl_xor(sum, off, 32);
  float inv = 1.0f / sum;
  for (int c = lane; c < cols; c += 32) o[c] = f2bf(p[c] * inv);
}

__global__ void gru_kernel(const float* __restrict__ gi, const float* __restrict__ gh,
                           float* __restrict__ h, bf16* __restrict__ hb, long n) {
  long i = (long)blockIdx.x * blockDim.x + threadIdx.x;
  if (i >= n) return;
  long row = i / D_; int c = (int)(i % D_);
  const float* gir = gi + row * 3 * D_;
  const float* ghr = gh + row * 3 * D_;
  float r = 1.f / (1.f + __expf(-(gir[c] + ghr[c])));
  float z = 1.f / (1.f + __expf(-(gir[D_ + c] + ghr[D_ + c])));
  float nn = tanhf(gir[2 * D_ + c] + r * ghr[2 * D_ + c]);
  float res = (1.f - z) * nn + z * h[i];
  h[i] = res; hb[i] = f2bf(res);
}

// ---------------------------------------------------------------------------
// Orchestration
// ---------------------------------------------------------------------------
extern "C" void kernel_launch(void* const* d_in, const int* in_sizes, int n_in,
                              void* d_out, int out_size, void* d_ws, size_t ws_size,
                              hipStream_t stream) {
  (void)in_sizes; (void)n_in; (void)out_size; (void)ws_size;

  const int*   tokens    = (const int*)  d_in[0];
  const float* mem       = (const float*)d_in[1];
  const float* cmem      = (const float*)d_in[2];
  const float* token_emb = (const float*)d_in[3];
  const float* pos_emb   = (const float*)d_in[4];
  const float* logits_w  = (const float*)d_in[5];
  const float* logits_b  = (const float*)d_in[6];
  const float* attn_ln_g = (const float*)d_in[7];
  const float* attn_ln_b = (const float*)d_in[8];
  const float* wq        = (const float*)d_in[9];
  const float* wkv       = (const float*)d_in[10];
  const float* wout      = (const float*)d_in[11];
  const float* bout      = (const float*)d_in[12];
  const float* conv_w    = (const float*)d_in[13];
  const float* conv_b    = (const float*)d_in[14];
  const float* a_wih     = (const float*)d_in[15];
  const float* a_whh     = (const float*)d_in[16];
  const float* a_bih     = (const float*)d_in[17];
  const float* a_bhh     = (const float*)d_in[18];
  const float* ff_ln_g   = (const float*)d_in[19];
  const float* ff_ln_b   = (const float*)d_in[20];
  const float* w1        = (const float*)d_in[21];
  const float* b1        = (const float*)d_in[22];
  const float* w2        = (const float*)d_in[23];
  const float* b2        = (const float*)d_in[24];
  const float* f_wih     = (const float*)d_in[25];
  const float* f_whh     = (const float*)d_in[26];
  const float* f_bih     = (const float*)d_in[27];
  const float* f_bhh     = (const float*)d_in[28];

  const long NTD = (long)B_ * T_ * D_;
  float* out        = (float*)d_out;
  float* out_logits = out;
  float* out_mems   = out + (long)B_ * T_ * V_;
  float* out_cmems  = out_mems + (long)DEPTH_ * NTD;

  // workspace carve (256B-aligned)
  char* wsb = (char*)d_ws;
  size_t off = 0;
  auto allocf = [&](long elems) -> float* {
    float* p = (float*)(wsb + off);
    off += ((size_t)elems * 4 + 255) & ~(size_t)255;
    return p;
  };
  auto allocb = [&](long elems) -> bf16* {
    bf16* p = (bf16*)(wsb + off);
    off += ((size_t)elems * 2 + 255) & ~(size_t)255;
    return p;
  };
  // fp32 state
  float* x    = allocf(NTD);
  float* dots = allocf((long)B_ * H_ * T_ * KV_);
  float* posr = allocf((long)B_ * H_ * T_ * KV_);
  float* gi   = allocf((long)B_ * T_ * 3 * D_);
  float* gh   = allocf((long)B_ * T_ * 3 * D_);
  // bf16 activations
  bf16* xb    = allocb(NTD);
  bf16* xnb   = allocb(NTD);
  bf16* ynb   = allocb(NTD);
  bf16* qb    = allocb(NTD);
  bf16* kvinb = allocb((long)B_ * KV_ * D_);
  bf16* kvb   = allocb((long)B_ * KV_ * 2 * D_);
  bf16* dotsb = allocb((long)B_ * H_ * T_ * KV_);
  bf16* vT    = allocb((long)B_ * H_ * DH_ * KV_);
  bf16* attnb = allocb(NTD);
  bf16* outpb = allocb(NTD);
  bf16* ffhb  = allocb((long)B_ * T_ * 4 * D_);
  bf16* memb  = allocb((long)B_ * MEM_LEN_ * D_);
  bf16* wpkb  = allocb((long)RATIO_ * D_ * D_);
  // bf16 weights, [N][K] layout
  bf16* wq_t  = allocb((long)DEPTH_ * D_ * D_);
  bf16* wkv_t = allocb((long)DEPTH_ * 2 * D_ * D_);
  bf16* wout_t= allocb((long)DEPTH_ * D_ * D_);
  bf16* w1_t  = allocb((long)DEPTH_ * 4 * D_ * D_);
  bf16* w2_t  = allocb((long)DEPTH_ * 4 * D_ * D_);
  bf16* lw_t  = allocb((long)V_ * D_);
  bf16* awih_b= allocb((long)DEPTH_ * 3 * D_ * D_);
  bf16* awhh_b= allocb((long)DEPTH_ * 3 * D_ * D_);
  bf16* fwih_b= allocb((long)DEPTH_ * 3 * D_ * D_);
  bf16* fwhh_b= allocb((long)DEPTH_ * 3 * D_ * D_);
  bf16* pos_b = allocb((long)H_ * KV_ * DH_);

  const float scale = 1.0f / sqrtf((float)DH_);
  const long dotsN  = (long)B_ * H_ * T_ * KV_;
  const long kvinN  = (long)B_ * KV_ * D_;
  const long vTN    = (long)B_ * H_ * DH_ * KV_;

  auto tgrid = [](int K, int N) { return dim3((N + 31) / 32, (K + 31) / 32, 1); };

  // ---- weight preparation (bf16, [N][K]) ----
  for (int l = 0; l < DEPTH_; ++l) {
    transpose_cvt_kernel<<<tgrid(D_, D_), 256, 0, stream>>>(
        wq + (long)l * D_ * D_, wq_t + (long)l * D_ * D_, D_, D_);
    transpose_cvt_kernel<<<tgrid(D_, 2 * D_), 256, 0, stream>>>(
        wkv + (long)l * D_ * 2 * D_, wkv_t + (long)l * 2 * D_ * D_, D_, 2 * D_);
    transpose_cvt_kernel<<<tgrid(D_, D_), 256, 0, stream>>>(
        wout + (long)l * D_ * D_, wout_t + (long)l * D_ * D_, D_, D_);
    transpose_cvt_kernel<<<tgrid(D_, 4 * D_), 256, 0, stream>>>(
        w1 + (long)l * D_ * 4 * D_, w1_t + (long)l * 4 * D_ * D_, D_, 4 * D_);
    transpose_cvt_kernel<<<tgrid(4 * D_, D_), 256, 0, stream>>>(
        w2 + (long)l * 4 * D_ * D_, w2_t + (long)l * D_ * 4 * D_, 4 * D_, D_);
  }
  transpose_cvt_kernel<<<tgrid(D_, V_), 256, 0, stream>>>(logits_w, lw_t, D_, V_);
  {
    long nw = (long)DEPTH_ * 3 * D_ * D_;
    cvt_kernel<<<(unsigned)((nw + 255) / 256), 256, 0, stream>>>(a_wih, awih_b, nw);
    cvt_kernel<<<(unsigned)((nw + 255) / 256), 256, 0, stream>>>(a_whh, awhh_b, nw);
    cvt_kernel<<<(unsigned)((nw + 255) / 256), 256, 0, stream>>>(f_wih, fwih_b, nw);
    cvt_kernel<<<(unsigned)((nw + 255) / 256), 256, 0, stream>>>(f_whh, fwhh_b, nw);
    long np = (long)H_ * KV_ * DH_;
    cvt_kernel<<<(unsigned)((np + 255) / 256), 256, 0, stream>>>(pos_emb, pos_b, np);
  }

  // x = token_emb[tokens]
  embed_kernel<<<(unsigned)((NTD + 255) / 256), 256, 0, stream>>>(tokens, token_emb, x, xb, NTD);

  for (int l = 0; l < DEPTH_; ++l) {
    const float* mem_l  = mem  + (long)l * B_ * MEM_LEN_ * D_;
    const float* cmem_l = cmem + (long)l * B_ * CMEM_LEN_ * D_;

    // ---- attention sublayer ----
    // LN writes fp32 straight into new_mems output (new_mems[l] == xn) + bf16 copy
    layernorm_kernel<<<B_ * T_, 256, 0, stream>>>(
        x, attn_ln_g + l * D_, attn_ln_b + l * D_, out_mems + (long)l * NTD, xnb, D_);

    launch_gemm<EPI_NONE, true>(stream, xnb, wq_t + (long)l * D_ * D_, qb, nullptr,
        B_ * T_, D_, D_, D_, D_, D_, 0, 0, 0, 0, 0, 0, 1, 1, 1.f);

    concat_kvin_kernel<<<(unsigned)((kvinN + 255) / 256), 256, 0, stream>>>(
        cmem_l, mem_l, xnb, kvinb, kvinN);

    launch_gemm<EPI_NONE, true>(stream, kvinb, wkv_t + (long)l * 2 * D_ * D_, kvb, nullptr,
        B_ * KV_, 2 * D_, D_, D_, D_, 2 * D_, 0, 0, 0, 0, 0, 0, 1, 1, 1.f);

    // scores: q[b,h] (512x128) x k[b,h]^T  (k slice of kv is already [j][dh])
    launch_gemm<EPI_SCALE, false>(stream, qb, kvb, dots, nullptr,
        T_, KV_, DH_, D_, 2 * D_, KV_,
        (long)T_ * D_, DH_, (long)KV_ * 2 * D_, DH_, (long)H_ * T_ * KV_, (long)T_ * KV_,
        B_ * H_, H_, scale);

    // positional scores: q[b,h] x pos_emb[h]^T
    launch_gemm<EPI_SCALE, false>(stream, qb, pos_b, posr, nullptr,
        T_, KV_, DH_, D_, DH_, KV_,
        (long)T_ * D_, DH_, 0, (long)KV_ * DH_, (long)H_ * T_ * KV_, (long)T_ * KV_,
        B_ * H_, H_, scale);

    shift_mask_add_kernel<<<(unsigned)((dotsN + 255) / 256), 256, 0, stream>>>(dots, posr, dotsN);
    softmax_rows_kernel<<<(unsigned)(((long)B_ * H_ * T_ + 7) / 8), 256, 0, stream>>>(
        dots, dotsb, (long)B_ * H_ * T_, KV_);

    // context: attn[b,h] (512x1152) x v[b,h]; V pre-transposed to [dh][j]
    vT_kernel<<<(unsigned)((vTN + 255) / 256), 256, 0, stream>>>(kvb, vT, vTN);
    launch_gemm<EPI_NONE, true>(stream, dotsb, vT, attnb, nullptr,
        T_, DH_, KV_, KV_, KV_, D_,
        (long)H_ * T_ * KV_, (long)T_ * KV_, (long)H_ * DH_ * KV_, (long)DH_ * KV_,
        (long)T_ * D_, DH_, B_ * H_, H_, 1.f);

    launch_gemm<EPI_BIAS, true>(stream, attnb, wout_t + (long)l * D_ * D_, outpb,
        bout + l * D_, B_ * T_, D_, D_, D_, D_, D_, 0, 0, 0, 0, 0, 0, 1, 1, 1.f);

    // attention GRU: x = GRU(outp, x)
    launch_gemm<EPI_BIAS, false>(stream, outpb, awih_b + (long)l * 3 * D_ * D_, gi,
        a_bih + (long)l * 3 * D_, B_ * T_, 3 * D_, D_, D_, D_, 3 * D_,
        0, 0, 0, 0, 0, 0, 1, 1, 1.f);
    launch_gemm<EPI_BIAS, false>(stream, xb, awhh_b + (long)l * 3 * D_ * D_, gh,
        a_bhh + (long)l * 3 * D_, B_ * T_, 3 * D_, D_, D_, D_, 3 * D_,
        0, 0, 0, 0, 0, 0, 1, 1, 1.f);
    gru_kernel<<<(unsigned)((NTD + 255) / 256), 256, 0, stream>>>(gi, gh, x, xb, NTD);

    // compressive memory: new_cmems[l] = conv1d(mem[l], stride 4) as GEMM
    {
      long nm = (long)B_ * MEM_LEN_ * D_;
      cvt_kernel<<<(unsigned)((nm + 255) / 256), 256, 0, stream>>>(mem_l, memb, nm);
      repack_conv_kernel<<<(unsigned)(((long)D_ * D_ * RATIO_ + 255) / 256), 256, 0, stream>>>(
          conv_w + (long)l * D_ * D_ * RATIO_, wpkb, (long)D_ * D_ * RATIO_);
      launch_gemm<EPI_BIAS, false>(stream, memb, wpkb,
          out_cmems + (long)l * B_ * CMEM_LEN_ * D_, conv_b + l * D_,
          CMEM_LEN_, D_, RATIO_ * D_, RATIO_ * D_, RATIO_ * D_, D_,
          (long)MEM_LEN_ * D_, 0, 0, 0, (long)CMEM_LEN_ * D_, 0, B_, 1, 1.f);
    }

    // ---- feed-forward sublayer ----
    layernorm_kernel<<<B_ * T_, 256, 0, stream>>>(
        x, ff_ln_g + l * D_, ff_ln_b + l * D_, (float*)nullptr, ynb, D_);
    launch_gemm<EPI_BIAS_GELU, true>(stream, ynb, w1_t + (long)l * 4 * D_ * D_, ffhb,
        b1 + (long)l * 4 * D_, B_ * T_, 4 * D_, D_, D_, D_, 4 * D_,
        0, 0, 0, 0, 0, 0, 1, 1, 1.f);
    launch_gemm<EPI_BIAS, true>(stream, ffhb, w2_t + (long)l * D_ * 4 * D_, outpb,
        b2 + l * D_, B_ * T_, D_, 4 * D_, 4 * D_, 4 * D_, D_,
        0, 0, 0, 0, 0, 0, 1, 1, 1.f);

    // FF GRU: x = GRU(outp, x)
    launch_gemm<EPI_BIAS, false>(stream, outpb, fwih_b + (long)l * 3 * D_ * D_, gi,
        f_bih + (long)l * 3 * D_, B_ * T_, 3 * D_, D_, D_, D_, 3 * D_,
        0, 0, 0, 0, 0, 0, 1, 1, 1.f);
    launch_gemm<EPI_BIAS, false>(stream, xb, fwhh_b + (long)l * 3 * D_ * D_, gh,
        f_bhh + (long)l * 3 * D_, B_ * T_, 3 * D_, D_, D_, D_, 3 * D_,
        0, 0, 0, 0, 0, 0, 1, 1, 1.f);
    gru_kernel<<<(unsigned)((NTD + 255) / 256), 256, 0, stream>>>(gi, gh, x, xb, NTD);
  }

  // logits = x @ logits_w + logits_b
  launch_gemm<EPI_BIAS, false>(stream, xb, lw_t, out_logits, logits_b,
      B_ * T_, V_, D_, D_, D_, V_, 0, 0, 0, 0, 0, 0, 1, 1, 1.f);
}
